// GeoMambaStudent_82471962018279
// MI455X (gfx1250) — compile-verified, matrix operand order checked by reading, not run on hardware
//
#include <hip/hip_runtime.h>
#include <math.h>

// ---------------- model dims ----------------
#define NLAY 2
#define NDIR 4
#define DD   256
#define DDI  512
#define DTRr 16
#define NST  16
#define KCV  4
#define BB   8
#define HH   16
#define WWd  16
#define LL   256          // HH*WWd
#define ROWS 2048         // BB*LL

typedef _Float16 half_t;
typedef __attribute__((ext_vector_type(16))) _Float16 v16h;
typedef __attribute__((ext_vector_type(8)))  _Float16 v8h;
typedef __attribute__((ext_vector_type(8)))  float    v8f;

// =====================================================================
// WMMA GEMM v2:  C[dir] = gather(A[dir]) @ Wh[dir]^T
//   A : M x K row-major f32 (converted to f16 on the fly)
//   Wh: N x K row-major f16 (pre-converted once per launch)
//   C : M x N f32
// Block = 128 threads (4 waves). Each wave owns 64 rows x 16 cols
// (4 accumulators); the whole block (256 rows) shares one 16x32 B tile
// staged into LDS per K-step via global_load_async_to_lds_b128 and
// consumed after s_wait_asynccnt + barrier.  gather==1 folds the
// direction flip permutation (bit0=flip W, bit1=flip H) into A rows.
// =====================================================================
__global__ __launch_bounds__(128) void k_gemm_wmma(
    const float* __restrict__ A, const _Float16* __restrict__ Wh,
    float* __restrict__ C, int M, int N, int K,
    long aDS, long wDS, long cDS, int gather)
{
  __shared__ _Float16 tileB[16 * 32];          // [n][k] row-major, 1 KB

  const int dir = blockIdx.z;
  const float*    Ad = A  + (gather ? 0L : (long)dir * aDS);
  const _Float16* Wd = Wh + (long)dir * wDS;
  float*          Cd = C  + (long)dir * cDS;

  const int tid  = threadIdx.x;
  const int wave = tid >> 5;
  const int lane = tid & 31;
  const int tm   = (blockIdx.x * 4 + wave) * 64;   // 64 rows per wave
  const int tn   = blockIdx.y * 16;
  const int ml   = lane & 15;
  const int hi   = lane >> 4;

  // A row pointers for the 4 M-subtiles (with optional flip gather)
  const float* ap[4];
  #pragma unroll
  for (int i = 0; i < 4; ++i) {
    int arow = tm + i * 16 + ml;
    if (gather) {
      int b = arow >> 8;
      int l = arow & 255;
      int h = l >> 4, w = l & 15;
      if (dir & 1) w = 15 - w;
      if (dir & 2) h = 15 - h;
      arow = b * LL + h * WWd + w;
    }
    ap[i] = Ad + (long)arow * K;
  }

  // staging: 64 chunks of 16B cover the 16x32 f16 B tile
  const int chunk = tid;                                   // valid if tid<64
  const _Float16* gsrc =
      Wd + (long)(tn + (chunk >> 2)) * K + (chunk & 3) * 8;
  const unsigned ldsOff =
      (unsigned)(size_t)(&tileB[(chunk & 63) * 8]);        // flat addr low32 = LDS offset

  v8f zero = {0.f,0.f,0.f,0.f,0.f,0.f,0.f,0.f};
  v8f acc[4] = {zero, zero, zero, zero};

  for (int k0 = 0; k0 < K; k0 += 32) {
    if (tid < 64) {
      unsigned long long ga = (unsigned long long)(gsrc + k0);
      asm volatile("global_load_async_to_lds_b128 %0, %1, off"
                   :: "v"(ldsOff), "v"(ga) : "memory");
    }
    asm volatile("s_wait_asynccnt 0x0" ::: "memory");
    __syncthreads();

    // B fragment from LDS (lane = column n, hi selects k half)
    v8h b0 = *(const v8h*)&tileB[ml * 32 + hi * 8];
    v8h b1 = *(const v8h*)&tileB[ml * 32 + 16 + hi * 8];
    v16h bf;
    #pragma unroll
    for (int t = 0; t < 8; ++t) { bf[t] = b0[t]; bf[t + 8] = b1[t]; }

    const int ka = k0 + hi * 8;
    #pragma unroll
    for (int i = 0; i < 4; ++i) {
      if (k0 + 32 < K) __builtin_prefetch(ap[i] + ka + 32, 0, 0);
      v16h af;
      #pragma unroll
      for (int t = 0; t < 8; ++t) {
        af[t]     = (half_t)ap[i][ka + t];
        af[t + 8] = (half_t)ap[i][ka + 16 + t];
      }
      acc[i] = __builtin_amdgcn_wmma_f32_16x16x32_f16(
          false, af, false, bf, (short)0, acc[i], false, false);
    }
    __syncthreads();                     // WAR: next slab overwrites tileB
  }

  #pragma unroll
  for (int i = 0; i < 4; ++i) {
    #pragma unroll
    for (int r = 0; r < 8; ++r) {        // C/D layout: m = r + 8*hi
      int m = tm + i * 16 + r + 8 * hi;
      Cd[(long)m * N + tn + ml] = acc[i][r];
    }
  }
}

// ---------------- f32 -> f16 weight conversion ----------------
__global__ void k_f2h(const float* __restrict__ src, _Float16* __restrict__ dst, int n)
{
  int i = blockIdx.x * 256 + threadIdx.x;
  if (i < n) dst[i] = (_Float16)src[i];
}

// ---------------- (B,C,H,W) -> (B, L, C) ----------------
__global__ void k_transpose(const float* __restrict__ feat, float* __restrict__ x)
{
  int i = blockIdx.x * 256 + threadIdx.x;       // B*L*D = 524288
  int c = i & 255, l = (i >> 8) & 255, b = i >> 16;
  int h = l >> 4, w = l & 15;
  x[i] = feat[((b * DD + c) * HH + h) * WWd + w];
}

// ---------------- causal depthwise conv (K=4) + SiLU ----------------
__global__ void k_conv(const float* __restrict__ xz,
                       const float* __restrict__ cw, const float* __restrict__ cb,
                       float* __restrict__ xconv)
{
  long i = (long)blockIdx.x * 256 + threadIdx.x;    // NDIR*B*L*DI = 4194304
  int di = (int)(i & 511);
  int l  = (int)((i >> 9) & 255);
  int br = (int)(i >> 17);                           // dir*8 + b
  int dir = br >> 3;
  const float* cwp = cw + ((long)dir * DDI + di) * KCV;
  float acc = cb[dir * DDI + di];
  long rb = (long)br * LL;
  #pragma unroll
  for (int k = 0; k < KCV; ++k) {
    int ls = l - 3 + k;
    if (ls >= 0) acc += xz[(rb + ls) * 1024 + di] * cwp[k];
  }
  xconv[i] = acc / (1.f + __expf(-acc));             // silu
}

// ---------------- windowed selective scan (K=8) ----------------
__global__ __launch_bounds__(128) void k_scan(
    const float* __restrict__ xdbl, const float* __restrict__ xconv,
    float* __restrict__ y,
    const float* __restrict__ dtw_, const float* __restrict__ dtb_,
    const float* __restrict__ alog_, const float* __restrict__ dvec_)
{
  int wave = threadIdx.x >> 5, lane = threadIdx.x & 31;
  int gw  = blockIdx.x * 4 + wave;                  // 8192 waves
  int dir = gw >> 11;
  int rem = gw & 2047;
  int b   = rem >> 8;
  int cp  = rem & 255;
  int hi  = lane >> 4, s = lane & 15;
  int di  = cp * 2 + hi;

  float dtw[DTRr];
  const float* dwp = dtw_ + ((long)(dir * DDI + di)) * DTRr;
  #pragma unroll
  for (int r = 0; r < DTRr; ++r) dtw[r] = dwp[r];
  float dtb = dtb_[dir * DDI + di];
  float Av  = -__expf(alog_[((long)(dir * DDI + di)) * NST + s]);
  float Dv  = dvec_[dir * DDI + di];

  float Cc = 0.f, ccb[7], dxb[7];
  #pragma unroll
  for (int j = 0; j < 7; ++j) { ccb[j] = 1e30f; dxb[j] = 0.f; }

  long seqb = (long)(dir * BB + b) * LL;
  for (int l = 0; l < LL; ++l) {
    const float* xd = xdbl + (seqb + l) * 48;
    float dv = dtb;
    #pragma unroll
    for (int r = 0; r < DTRr; ++r) dv += xd[r] * dtw[r];
    float dt = (dv > 20.f) ? dv : log1pf(__expf(dv));   // softplus
    float Bp = xd[16 + s], Cp = xd[32 + s];
    float xc = xconv[(seqb + l) * DDI + di];
    Cc += dt * Av;
    float dbx = dt * Bp * xc;
    float h = dbx;
    #pragma unroll
    for (int j = 0; j < 7; ++j) h += __expf(Cc - ccb[j]) * dxb[j];
    #pragma unroll
    for (int j = 6; j > 0; --j) { ccb[j] = ccb[j-1]; dxb[j] = dxb[j-1]; }
    ccb[0] = Cc; dxb[0] = dbx;
    float yl = h * Cp;                               // reduce over 16 states
    yl += __shfl_xor(yl, 1, 32);
    yl += __shfl_xor(yl, 2, 32);
    yl += __shfl_xor(yl, 4, 32);
    yl += __shfl_xor(yl, 8, 32);
    if (s == 0) y[(seqb + l) * DDI + di] = yl + Dv * xc;
  }
}

// ---------------- y *= silu(z) ----------------
__global__ void k_ymul(float* __restrict__ y, const float* __restrict__ xz)
{
  long i = (long)blockIdx.x * 256 + threadIdx.x;    // NDIR*ROWS*DI
  int di = (int)(i & 511);
  long row = i >> 9;                                 // dir*ROWS + r
  float z = xz[row * 1024 + DDI + di];
  y[i] *= z / (1.f + __expf(-z));
}

// ------ residual + LayerNorm + scatter into combined (unflip) ------
__global__ __launch_bounds__(128) void k_ln_comb(
    const float* __restrict__ o, const float* __restrict__ xin,
    const float* __restrict__ ng, const float* __restrict__ nb,
    float* __restrict__ comb)
{
  int wave = threadIdx.x >> 5, lane = threadIdx.x & 31;
  int gw = blockIdx.x * 4 + wave;                   // 8192 rows
  int dir = gw >> 11, row = gw & 2047;
  int b = row >> 8, l = row & 255;
  int h = l >> 4, w = l & 15;
  if (dir & 1) w = 15 - w;
  if (dir & 2) h = 15 - h;
  int lo = h * WWd + w;                              // original position
  const float* op = o + ((long)dir * ROWS + row) * DD;
  const float* xp = xin + ((long)b * LL + lo) * DD;
  float vals[8], sum = 0.f;
  #pragma unroll
  for (int j = 0; j < 8; ++j) { int c = lane + 32*j; float v = op[c] + xp[c]; vals[j] = v; sum += v; }
  for (int off = 16; off; off >>= 1) sum += __shfl_xor(sum, off, 32);
  float m = sum * (1.f / 256.f), s2 = 0.f;
  #pragma unroll
  for (int j = 0; j < 8; ++j) { float d = vals[j] - m; s2 += d * d; }
  for (int off = 16; off; off >>= 1) s2 += __shfl_xor(s2, off, 32);
  float rstd = rsqrtf(s2 * (1.f / 256.f) + 1e-5f);
  float* cp = comb + ((long)b * LL + lo) * (4 * DD) + dir * DD;
  #pragma unroll
  for (int j = 0; j < 8; ++j) {
    int c = lane + 32 * j;
    cp[c] = (vals[j] - m) * rstd * ng[dir * DD + c] + nb[dir * DD + c];
  }
}

// ---------------- bias + exact GELU ----------------
__global__ void k_biasgelu(float* __restrict__ hfu, const float* __restrict__ b1)
{
  long i = (long)blockIdx.x * 256 + threadIdx.x;    // ROWS*512
  int c = (int)(i & 511);
  float v = hfu[i] + b1[c];
  hfu[i] = 0.5f * v * (1.f + erff(v * 0.70710678118654752f));
}

// ---------------- altitude gate ----------------
__global__ void k_gate(const float* __restrict__ alt_emb, const int* __restrict__ alt_idx,
                       const float* __restrict__ gw, const float* __restrict__ gb,
                       float* __restrict__ gate)
{
  int i = blockIdx.x * 256 + threadIdx.x;           // B*D = 2048
  int b = i >> 8, c = i & 255;
  const float* e = alt_emb + alt_idx[b] * 32;
  const float* wr = gw + (long)c * 32;
  float acc = gb[c];
  #pragma unroll
  for (int j = 0; j < 32; ++j) acc += e[j] * wr[j];
  gate[i] = 1.f / (1.f + __expf(-acc));
}

// ---------------- bias + LayerNorm + gate multiply ----------------
__global__ __launch_bounds__(128) void k_ln_gate(
    const float* __restrict__ t2, const float* __restrict__ b2,
    const float* __restrict__ lg, const float* __restrict__ lb,
    const float* __restrict__ gate, float* __restrict__ xout)
{
  int wave = threadIdx.x >> 5, lane = threadIdx.x & 31;
  int row = blockIdx.x * 4 + wave;                  // ROWS
  int b = row >> 8;
  const float* tp = t2 + (long)row * DD;
  float vals[8], sum = 0.f;
  #pragma unroll
  for (int j = 0; j < 8; ++j) { int c = lane + 32*j; float v = tp[c] + b2[c]; vals[j] = v; sum += v; }
  for (int off = 16; off; off >>= 1) sum += __shfl_xor(sum, off, 32);
  float m = sum * (1.f / 256.f), s2 = 0.f;
  #pragma unroll
  for (int j = 0; j < 8; ++j) { float d = vals[j] - m; s2 += d * d; }
  for (int off = 16; off; off >>= 1) s2 += __shfl_xor(s2, off, 32);
  float rstd = rsqrtf(s2 * (1.f / 256.f) + 1e-5f);
  #pragma unroll
  for (int j = 0; j < 8; ++j) {
    int c = lane + 32 * j;
    xout[(long)row * DD + c] =
        ((vals[j] - m) * rstd * lg[c] + lb[c]) * gate[b * DD + c];
  }
}

// =====================================================================
extern "C" void kernel_launch(void* const* d_in, const int* in_sizes, int n_in,
                              void* d_out, int out_size, void* d_ws, size_t ws_size,
                              hipStream_t stream)
{
  (void)in_sizes; (void)n_in; (void)out_size;
  const float* feat    = (const float*)d_in[0];
  const int*   alt_idx = (const int*)  d_in[1];
  const float* in_w    = (const float*)d_in[2];
  const float* dt_w    = (const float*)d_in[3];
  const float* dt_b    = (const float*)d_in[4];
  const float* A_log   = (const float*)d_in[5];
  const float* Dvec    = (const float*)d_in[6];
  const float* xp_w    = (const float*)d_in[7];
  const float* cw      = (const float*)d_in[8];
  const float* cb      = (const float*)d_in[9];
  const float* out_w   = (const float*)d_in[10];
  const float* ng      = (const float*)d_in[11];
  const float* nb      = (const float*)d_in[12];
  const float* f_w1    = (const float*)d_in[13];
  const float* f_b1    = (const float*)d_in[14];
  const float* f_w2    = (const float*)d_in[15];
  const float* f_b2    = (const float*)d_in[16];
  const float* f_lg    = (const float*)d_in[17];
  const float* f_lb    = (const float*)d_in[18];
  const float* alt_emb = (const float*)d_in[19];
  const float* g_w     = (const float*)d_in[20];
  const float* g_b     = (const float*)d_in[21];

  float* ws = (float*)d_ws;
  size_t o_x0 = 0;
  size_t o_x1 = o_x0 + (size_t)ROWS * DD;
  size_t o_xz = o_x1 + (size_t)ROWS * DD;
  size_t o_xc = o_xz + (size_t)NDIR * ROWS * 1024;
  size_t o_xd = o_xc + (size_t)NDIR * ROWS * DDI;
  size_t o_y  = o_xd + (size_t)NDIR * ROWS * 48;
  size_t o_o  = o_y  + (size_t)NDIR * ROWS * DDI;
  size_t o_cm = o_o  + (size_t)NDIR * ROWS * DD;
  size_t o_hf = o_cm + (size_t)ROWS * 4 * DD;
  size_t o_t2 = o_hf + (size_t)ROWS * 512;
  size_t o_g  = o_t2 + (size_t)ROWS * DD;
  size_t o_wh = o_g  + (size_t)BB * DD;            // f16 weight pool (in float units)

  // f16 weight pool sub-offsets (in halves)
  const size_t n_inw = (size_t)NLAY * NDIR * 1024 * DD;   // 2097152
  const size_t n_xpw = (size_t)NLAY * NDIR * 48 * DDI;    // 196608
  const size_t n_ouw = (size_t)NLAY * NDIR * DD * DDI;    // 1048576
  const size_t n_fw1 = (size_t)NLAY * 512 * 1024;         // 1048576
  const size_t n_fw2 = (size_t)NLAY * DD * 512;           // 262144
  const size_t h_in  = 0;
  const size_t h_xp  = h_in + n_inw;
  const size_t h_ou  = h_xp + n_xpw;
  const size_t h_f1  = h_ou + n_ouw;
  const size_t h_f2  = h_f1 + n_fw1;
  const size_t h_tot = h_f2 + n_fw2;                      // 4653056 halves
  size_t total = o_wh + (h_tot + 1) / 2;
  if (ws_size < total * sizeof(float)) return;

  float* xz   = ws + o_xz;
  float* xcv  = ws + o_xc;
  float* xdb  = ws + o_xd;
  float* yb   = ws + o_y;
  float* ob   = ws + o_o;
  float* comb = ws + o_cm;
  float* hfu  = ws + o_hf;
  float* t2   = ws + o_t2;
  float* gbuf = ws + o_g;
  _Float16* wh = (_Float16*)(ws + o_wh);

  float* xcur  = ws + o_x0;
  float* xnext = ws + o_x1;

  // one-time f32 -> f16 weight conversion (covers both layers)
  k_f2h<<<(int)((n_inw + 255) / 256), 256, 0, stream>>>(in_w,  wh + h_in, (int)n_inw);
  k_f2h<<<(int)((n_xpw + 255) / 256), 256, 0, stream>>>(xp_w,  wh + h_xp, (int)n_xpw);
  k_f2h<<<(int)((n_ouw + 255) / 256), 256, 0, stream>>>(out_w, wh + h_ou, (int)n_ouw);
  k_f2h<<<(int)((n_fw1 + 255) / 256), 256, 0, stream>>>(f_w1,  wh + h_f1, (int)n_fw1);
  k_f2h<<<(int)((n_fw2 + 255) / 256), 256, 0, stream>>>(f_w2,  wh + h_f2, (int)n_fw2);

  k_transpose<<<2048, 256, 0, stream>>>(feat, xcur);

  for (int li = 0; li < NLAY; ++li) {
    const _Float16* inw_l  = wh + h_in + (size_t)li * NDIR * 1024 * DD;
    const _Float16* xpw_l  = wh + h_xp + (size_t)li * NDIR * 48 * DDI;
    const _Float16* outw_l = wh + h_ou + (size_t)li * NDIR * DD * DDI;
    const _Float16* fw1_l  = wh + h_f1 + (size_t)li * 512 * 1024;
    const _Float16* fw2_l  = wh + h_f2 + (size_t)li * DD * 512;
    const float* dtw_l  = dt_w  + (size_t)li * NDIR * DDI * DTRr;
    const float* dtb_l  = dt_b  + (size_t)li * NDIR * DDI;
    const float* alog_l = A_log + (size_t)li * NDIR * DDI * NST;
    const float* d_l    = Dvec  + (size_t)li * NDIR * DDI;
    const float* cw_l   = cw    + (size_t)li * NDIR * DDI * KCV;
    const float* cb_l   = cb    + (size_t)li * NDIR * DDI;
    const float* ng_l   = ng    + (size_t)li * NDIR * DD;
    const float* nb_l   = nb    + (size_t)li * NDIR * DD;
    const float* fb1_l  = f_b1  + (size_t)li * 512;
    const float* fb2_l  = f_b2  + (size_t)li * DD;
    const float* flg_l  = f_lg  + (size_t)li * DD;
    const float* flb_l  = f_lb  + (size_t)li * DD;
    const float* gw_l   = g_w   + (size_t)li * DD * 32;
    const float* gb_l   = g_b   + (size_t)li * DD;

    // xz[d] = gather(x, perm_d) @ in_w[d]^T       (2048 x 1024, K=256)
    k_gemm_wmma<<<dim3(8, 64, 4), 128, 0, stream>>>(
        xcur, inw_l, xz, ROWS, 1024, DD, 0L, 1024L * DD, (long)ROWS * 1024, 1);
    // depthwise conv + silu
    k_conv<<<16384, 256, 0, stream>>>(xz, cw_l, cb_l, xcv);
    // x_dbl = x_conv @ xp_w^T                     (2048 x 48, K=512)
    k_gemm_wmma<<<dim3(8, 3, 4), 128, 0, stream>>>(
        xcv, xpw_l, xdb, ROWS, 48, DDI, (long)ROWS * DDI, 48L * DDI, (long)ROWS * 48, 0);
    // selective scan (dt on the fly)
    k_scan<<<2048, 128, 0, stream>>>(xdb, xcv, yb, dtw_l, dtb_l, alog_l, d_l);
    // y *= silu(z)
    k_ymul<<<16384, 256, 0, stream>>>(yb, xz);
    // o = y @ out_w^T                             (2048 x 256, K=512)
    k_gemm_wmma<<<dim3(8, 16, 4), 128, 0, stream>>>(
        yb, outw_l, ob, ROWS, DD, DDI, (long)ROWS * DDI, (long)DD * DDI, (long)ROWS * DD, 0);
    // residual + LN + scatter (unflip) into combined
    k_ln_comb<<<2048, 128, 0, stream>>>(ob, xcur, ng_l, nb_l, comb);
    // hfu = combined @ fus_w1^T                   (2048 x 512, K=1024)
    k_gemm_wmma<<<dim3(8, 32, 1), 128, 0, stream>>>(
        comb, fw1_l, hfu, ROWS, 512, 1024, 0L, 0L, 0L, 0);
    k_biasgelu<<<4096, 256, 0, stream>>>(hfu, fb1_l);
    // t2 = hfu @ fus_w2^T                         (2048 x 256, K=512)
    k_gemm_wmma<<<dim3(8, 16, 1), 128, 0, stream>>>(
        hfu, fw2_l, t2, ROWS, DD, 512, 0L, 0L, 0L, 0);
    // gate
    k_gate<<<8, 256, 0, stream>>>(alt_emb, alt_idx, gw_l, gb_l, gbuf);
    // bias + LN + gate -> next x (last layer writes d_out)
    float* xo = (li == NLAY - 1) ? (float*)d_out : xnext;
    k_ln_gate<<<512, 128, 0, stream>>>(t2, fb2_l, flg_l, flb_l, gbuf, xo);
    xcur = xo;
  }
}
